// EdgeConvBlock_17145509446483
// MI455X (gfx1250) — matrix operand order repaired and checked by reading, not compiled, not run on hardware
//
#include <hip/hip_runtime.h>

typedef _Float16 v16h __attribute__((ext_vector_type(16)));
typedef _Float16 v8h  __attribute__((ext_vector_type(8)));
typedef float    v8f  __attribute__((ext_vector_type(8)));

#define KNN      32
#define NPTS     4096
#define NBATCH   8
#define PTS_BLK  128
#define THREADS  128
#define CHUNK_P  16
#define H1_PAD   72   // 64 f16 channels + 8 pad -> 144B row stride (16B aligned, bank-spread)

__global__ __launch_bounds__(THREADS)
void edgeconv_fused_kernel(const float* __restrict__ x,
                           const float* __restrict__ w1,
                           const float* __restrict__ g1, const float* __restrict__ b1,
                           const float* __restrict__ m1, const float* __restrict__ v1,
                           const float* __restrict__ w2,
                           const float* __restrict__ g2, const float* __restrict__ b2,
                           const float* __restrict__ m2, const float* __restrict__ v2,
                           float* __restrict__ out)
{
    __shared__ float     xs[3 * NPTS];                      // 48 KB  point cloud (SoA)
    __shared__ _Float16  h1s[32 * CHUNK_P * H1_PAD];        // 72 KB  conv1 output chunk [k][p][ch]
    __shared__ int       idxs[PTS_BLK * KNN];               // 16 KB  neighbor indices
    __shared__ float     wf[64 * 6];                        // BN1-folded conv1 weights
    __shared__ float     bfold[64];                         // BN1-folded bias
    __shared__ float     sh2[64];                           // BN2 shift
    __shared__ _Float16  w2h[64 * 64];                      // 8 KB   BN2-scaled conv2 weights (f16)

    const int tid = threadIdx.x;
    const int bb  = blockIdx.y;
    const int n0  = blockIdx.x * PTS_BLK;

    // ---- Stage A0: fold batchnorms into weights ----
    if (tid < 64) {
        float s1 = g1[tid] * rsqrtf(v1[tid] + 1e-5f);
        #pragma unroll
        for (int c = 0; c < 6; ++c) wf[tid * 6 + c] = w1[tid * 6 + c] * s1;
        bfold[tid] = b1[tid] - m1[tid] * s1;
        float s2 = g2[tid] * rsqrtf(v2[tid] + 1e-5f);
        sh2[tid]  = b2[tid] - m2[tid] * s2;
        for (int c = 0; c < 64; ++c)
            w2h[tid * 64 + c] = (_Float16)(w2[tid * 64 + c] * s2);
    }

    // ---- Stage A1: stage this batch's point cloud into LDS ----
    const float* xb = x + (size_t)bb * 3 * NPTS;
    for (int i = tid; i < 3 * NPTS; i += THREADS) xs[i] = xb[i];
    __syncthreads();

    // ---- Stage B: exact kNN (top-32 smallest sq-dist) per thread ----
    const int   n  = n0 + tid;
    const float cx = xs[n], cy = xs[NPTS + n], cz = xs[2 * NPTS + n];
    float dk[KNN]; int ik[KNN];
    #pragma unroll
    for (int j = 0; j < KNN; ++j) { dk[j] = 3.0e38f; ik[j] = 0; }
    float dmax = 3.0e38f; int pmax = 0;
    for (int m = 0; m < NPTS; ++m) {
        float dx = xs[m]            - cx;
        float dy = xs[NPTS + m]     - cy;
        float dz = xs[2 * NPTS + m] - cz;
        float dist = dx * dx + dy * dy + dz * dz;
        if (dist < dmax) {
            #pragma unroll
            for (int j = 0; j < KNN; ++j) if (j == pmax) { dk[j] = dist; ik[j] = m; }
            dmax = dk[0]; pmax = 0;
            #pragma unroll
            for (int j = 1; j < KNN; ++j) if (dk[j] > dmax) { dmax = dk[j]; pmax = j; }
        }
    }
    #pragma unroll
    for (int j = 0; j < KNN; ++j) idxs[tid * KNN + j] = ik[j];
    __syncthreads();

    // ---- Load per-wave A fragments of conv2 weights (Mtile = wave id) ----
    const int wv = tid >> 5, lane = tid & 31, halfw = lane >> 4, mrow = lane & 15;
    const int Mrow = wv * 16 + mrow;
    v16h a0, a1;
    {
        // 16-bit A-matrix 16x32 layout: e=0..7 -> K = 8*half + e ; e=8..15 -> K = 16 + 8*half + (e-8)
        v8h l0 = *(const v8h*)&w2h[Mrow * 64 +  0 + 8 * halfw];
        v8h h0 = *(const v8h*)&w2h[Mrow * 64 + 16 + 8 * halfw];
        v8h l1 = *(const v8h*)&w2h[Mrow * 64 + 32 + 8 * halfw];
        v8h h1v= *(const v8h*)&w2h[Mrow * 64 + 48 + 8 * halfw];
        #pragma unroll
        for (int e = 0; e < 8; ++e) {
            a0[e] = l0[e]; a0[e + 8] = h0[e];
            a1[e] = l1[e]; a1[e + 8] = h1v[e];
        }
    }

    const size_t outbase = (size_t)bb * 64 * NPTS;

    // ---- Stage C/D: per-chunk conv1 -> LDS(f16) -> WMMA conv2 -> k-max ----
    for (int ch = 0; ch < PTS_BLK / CHUNK_P; ++ch) {
        // conv1 + BN1 + lrelu for 16 points x 32 k  (512 jobs / 128 threads)
        #pragma unroll 1
        for (int job = 0; job < 4; ++job) {
            int j  = tid + job * THREADS;
            int p  = j & 15, kk = j >> 4;
            int pg = ch * CHUNK_P + p;
            int nb = idxs[pg * KNN + kk];
            int np = n0 + pg;
            float f0 = xs[np], f1 = xs[NPTS + np], f2 = xs[2 * NPTS + np];
            float f3 = xs[nb]            - f0;
            float f4 = xs[NPTS + nb]     - f1;
            float f5 = xs[2 * NPTS + nb] - f2;
            unsigned int* row = (unsigned int*)&h1s[(kk * CHUNK_P + p) * H1_PAD];
            #pragma unroll 8
            for (int o = 0; o < 64; o += 2) {
                float y0 = bfold[o]     + wf[o*6+0]*f0 + wf[o*6+1]*f1 + wf[o*6+2]*f2
                                        + wf[o*6+3]*f3 + wf[o*6+4]*f4 + wf[o*6+5]*f5;
                float y1 = bfold[o+1]   + wf[o*6+6]*f0 + wf[o*6+7]*f1 + wf[o*6+8]*f2
                                        + wf[o*6+9]*f3 + wf[o*6+10]*f4 + wf[o*6+11]*f5;
                // lrelu(y) == max(y, 0.2*y) since 0 < slope < 1  -> pure v_max, no cmp/cndmask
                y0 = fmaxf(y0, 0.2f * y0);
                y1 = fmaxf(y1, 0.2f * y1);
                auto pk = __builtin_amdgcn_cvt_pkrtz(y0, y1);  // v_cvt_pk_rtz_f16_f32
                row[o >> 1] = __builtin_bit_cast(unsigned int, pk);
            }
        }
        __syncthreads();

        // WMMA: wave wv computes out-channels [16*wv, 16*wv+16) for the 16 points
        v8f mx;
        #pragma unroll
        for (int v = 0; v < 8; ++v) mx[v] = -3.0e38f;

        #pragma unroll 2
        for (int kt = 0; kt < 32; ++kt) {
            // 16-bit B-matrix 32x16: lane n = lane&15, e=0..7 -> K = 8*half + e ; e=8..15 -> K = 16 + 8*half + (e-8)
            const _Float16* rb = &h1s[(kt * CHUNK_P + mrow) * H1_PAD];
            v8h bl0 = *(const v8h*)&rb[ 0 + 8 * halfw];
            v8h bh0 = *(const v8h*)&rb[16 + 8 * halfw];
            v8h bl1 = *(const v8h*)&rb[32 + 8 * halfw];
            v8h bh1 = *(const v8h*)&rb[48 + 8 * halfw];
            v16h bb0, bb1;
            #pragma unroll
            for (int e = 0; e < 8; ++e) {
                bb0[e] = bl0[e]; bb0[e + 8] = bh0[e];
                bb1[e] = bl1[e]; bb1[e + 8] = bh1[e];
            }
            v8f acc = {};
            acc = __builtin_amdgcn_wmma_f32_16x16x32_f16(false, a0, false, bb0,
                                                         (short)0, acc, false, false);
            acc = __builtin_amdgcn_wmma_f32_16x16x32_f16(false, a1, false, bb1,
                                                         (short)0, acc, false, false);
            #pragma unroll
            for (int v = 0; v < 8; ++v) {
                float ys = acc[v] + sh2[wv * 16 + v + 8 * halfw];
                // lrelu + running k-max as a pure max tree
                mx[v] = fmaxf(mx[v], fmaxf(ys, 0.2f * ys));
            }
        }

        // store k-max: lanes 0-15 cover 16 consecutive points -> coalesced
        int np = n0 + ch * CHUNK_P + mrow;
        #pragma unroll
        for (int v = 0; v < 8; ++v) {
            int oc = wv * 16 + v + 8 * halfw;
            out[outbase + (size_t)oc * NPTS + np] = mx[v];
        }
        __syncthreads();
    }
}

extern "C" void kernel_launch(void* const* d_in, const int* in_sizes, int n_in,
                              void* d_out, int out_size, void* d_ws, size_t ws_size,
                              hipStream_t stream) {
    (void)in_sizes; (void)n_in; (void)d_ws; (void)ws_size; (void)out_size;
    dim3 grid(NPTS / PTS_BLK, NBATCH);
    edgeconv_fused_kernel<<<grid, THREADS, 0, stream>>>(
        (const float*)d_in[0],  // x
        (const float*)d_in[1],  // w1
        (const float*)d_in[2], (const float*)d_in[3],   // gamma1, beta1
        (const float*)d_in[4], (const float*)d_in[5],   // mean1, var1
        (const float*)d_in[6],  // w2
        (const float*)d_in[7], (const float*)d_in[8],   // gamma2, beta2
        (const float*)d_in[9], (const float*)d_in[10],  // mean2, var2
        (float*)d_out);
}
